// MotionEncoderGCN_88768384073996
// MI455X (gfx1250) — matrix operand order repaired and checked by reading, not compiled, not run on hardware
//
#include <hip/hip_runtime.h>
#include <hip/hip_bf16.h>

// ---------------------------------------------------------------------------
// MotionEncoderGCN forward for MI455X (gfx1250, wave32, WMMA).
//  - LSTM recurrence: v_wmma_f32_16x16x32_bf16, weights pre-swizzled in LDS
//  - GCN linear:      v_wmma_f32_16x16x4_f32, W pre-swizzled in LDS so each
//                     B operand is a single contiguous ds_load_b64
//  - Aggregation:     native f32 global atomics (whole working set fits L2)
// ---------------------------------------------------------------------------

typedef __attribute__((ext_vector_type(16))) __bf16 v16bf;
typedef __attribute__((ext_vector_type(8)))  float  v8f;
typedef __attribute__((ext_vector_type(2)))  float  v2f;

#define A_TOT  20000
#define L_TOT  80000
#define N_TOT  100000
#define HDIM   64
#define E_AA_N 400000
#define E_AL_N 800000
#define BN_EPS 1e-5f

union V16U { v16bf v; uint4 q[2]; };

__device__ __forceinline__ float sigf(float x)     { return 1.f / (1.f + __expf(-x)); }
__device__ __forceinline__ float tanhfast(float x) { return 2.f * sigf(2.f * x) - 1.f; }

__device__ __forceinline__ v8f wmma_bf16(v16bf a, v16bf b, v8f c) {
  return __builtin_amdgcn_wmma_f32_16x16x32_bf16(false, a, false, b, (short)0, c, false, false);
}
__device__ __forceinline__ v8f wmma_f32x4(v2f a, v2f b, v8f c) {
  return __builtin_amdgcn_wmma_f32_16x16x4_f32(false, a, false, b, (short)0, c, false, false);
}

// ---------------------------------------------------------------------------
// LSTM: one wave = 16 rows. Gates = [x_t | h] @ [Wih | Whh].T + b, K=96 as
// 3 bf16 K-chunks of 32. Weights swizzled once into LDS in a contiguous-K
// B-fragment layout; h round-trips through a padded LDS tile (row stride
// 144B -> 16 rows hit 16 distinct bank groups) to convert the C/D layout
// into the A-fragment layout. LDS ops are in-order per wave -> no barrier
// inside the recurrence. LDS: 48K (B-frags) + 18K (h tiles) = 67584 B.
// ---------------------------------------------------------------------------
template<int T, int D, bool MEAN>
__global__ void __launch_bounds__(256)
lstm_kernel(const float* __restrict__ X, const float* __restrict__ Wih,
            const float* __restrict__ Whh, const float* __restrict__ bvec,
            float* __restrict__ emb_out, float* __restrict__ xfeat_out, int Nrows)
{
  extern __shared__ char smem[];
  __bf16* bfw = (__bf16*)smem;                   // [16 tiles][3 chunks][32 lanes][16]
  __bf16* hb  = (__bf16*)(smem + 49152);         // 8 waves * 16 rows * 72 (padded)

  const int tid = threadIdx.x;
  // Swizzled B-fragments of Wz = [Whh | Wih | 0]  (B elem e of lane l:
  // K = 32c + 16*(l>>4) + e, gate col = 16t + (l&15)).
  for (int fi = tid; fi < 24576; fi += 256) {
    int e = fi & 15, l = (fi >> 4) & 31, tc = fi >> 9;
    int c = tc % 3, t = tc / 3;
    int K   = 32 * c + 16 * (l >> 4) + e;
    int col = 16 * t + (l & 15);
    float v = 0.f;
    if (K < HDIM)          v = Whh[col * HDIM + K];
    else if (K - HDIM < D) v = Wih[col * D + (K - HDIM)];
    bfw[fi] = (__bf16)v;
  }
  __syncthreads();

  const int lane = tid & 31, wave = tid >> 5;
  const int half = lane >> 4, ncol = lane & 15;
  const int rowbase = blockIdx.x * 128 + wave * 16;
  const int myrow = rowbase + ncol;
  const int xrow  = (myrow < Nrows) ? myrow : (Nrows - 1);

  const v16bf* bv = (const v16bf*)bfw;           // 32B-aligned fragment loads
  const uint4* hq = (const uint4*)hb;

  // Per-lane bias registers: biasr[gate][q] for column ncol of tile q.
  float biasr[4][4];
  #pragma unroll
  for (int g = 0; g < 4; ++g) {
    #pragma unroll
    for (int q = 0; q < 4; ++q) biasr[g][q] = bvec[g * 64 + q * 16 + ncol];
  }

  float cst[4][8], hs[4][8];
  #pragma unroll
  for (int q = 0; q < 4; ++q) {
    #pragma unroll
    for (int e = 0; e < 8; ++e) { cst[q][e] = 0.f; hs[q][e] = 0.f; }
  }

  v16bf a0 = {}; v16bf a1 = {};   // h A-fragments (h starts at 0)

  #pragma unroll 1
  for (int s = 0; s < T; ++s) {
    // x_t chunk (K=64..95): only K<64+D nonzero -> half-0 lanes, elems 0..D-1
    v16bf a2 = {};
    if (half == 0) {
      const float* xp = X + ((size_t)xrow * T + s) * D;
      #pragma unroll
      for (int d = 0; d < D; ++d) a2[d] = (__bf16)xp[d];
    }

    #pragma unroll
    for (int q = 0; q < 4; ++q) {         // gate col tiles: i=q f=q+4 g=q+8 o=q+12
      float bi0 = biasr[0][q], bf0 = biasr[1][q];
      float bg0 = biasr[2][q], bo0 = biasr[3][q];
      v8f ai = {bi0,bi0,bi0,bi0,bi0,bi0,bi0,bi0};
      v8f af = {bf0,bf0,bf0,bf0,bf0,bf0,bf0,bf0};
      v8f ag = {bg0,bg0,bg0,bg0,bg0,bg0,bg0,bg0};
      v8f ao = {bo0,bo0,bo0,bo0,bo0,bo0,bo0,bo0};

      #pragma unroll
      for (int c = 0; c < 3; ++c) {
        v16bf av = (c == 0) ? a0 : ((c == 1) ? a1 : a2);
        v16bf wbi = bv[((q     ) * 3 + c) * 32 + lane];
        v16bf wbf = bv[((q +  4) * 3 + c) * 32 + lane];
        v16bf wbg = bv[((q +  8) * 3 + c) * 32 + lane];
        v16bf wbo = bv[((q + 12) * 3 + c) * 32 + lane];
        ai = wmma_bf16(av, wbi, ai);
        af = wmma_bf16(av, wbf, af);
        ag = wmma_bf16(av, wbg, ag);
        ao = wmma_bf16(av, wbo, ao);
      }

      #pragma unroll
      for (int e = 0; e < 8; ++e) {       // C/D layout: row = e + 8*half, col = ncol
        float iv = sigf(ai[e]), fv = sigf(af[e]);
        float gv = tanhfast(ag[e]), ov = sigf(ao[e]);
        float cc = fv * cst[q][e] + iv * gv;
        cst[q][e] = cc;
        float hv = ov * tanhfast(cc);
        hs[q][e] = MEAN ? (hs[q][e] + hv) : hv;
        hb[wave * 1152 + (e + 8 * half) * 72 + q * 16 + ncol] = (__bf16)hv;
      }
    }

    // Rebuild h A-fragments for next step (LDS ops are in-order per wave).
    {
      V16U u0, u1;
      int i0 = wave * 144 + ncol * 9 + 2 * half;       // chunk 0
      u0.q[0] = hq[i0];     u0.q[1] = hq[i0 + 1];
      int i1 = wave * 144 + ncol * 9 + 4 + 2 * half;   // chunk 1
      u1.q[0] = hq[i1];     u1.q[1] = hq[i1 + 1];
      a0 = u0.v; a1 = u1.v;
    }
  }

  const float sc = MEAN ? (1.f / (float)T) : 1.f;
  #pragma unroll
  for (int q = 0; q < 4; ++q) {
    #pragma unroll
    for (int e = 0; e < 8; ++e) {
      int r = rowbase + e + 8 * half;
      if (r < Nrows) {
        float v = hs[q][e] * sc;
        emb_out  [(size_t)r * HDIM + q * 16 + ncol] = v;
        xfeat_out[(size_t)r * HDIM + q * 16 + ncol] = v;
      }
    }
  }
}

// ---------------------------------------------------------------------------
// Exact-f32 node linear: Out = X @ W via v_wmma_f32_16x16x4_f32.
// W is pre-swizzled into LDS in B-fragment layout, so each B operand is one
// contiguous 8-byte ds load landing directly in an aligned VGPR pair.
// ---------------------------------------------------------------------------
__global__ void __launch_bounds__(256)
linear_kernel(const float* __restrict__ Xin, const float* __restrict__ W,
              float* __restrict__ Out, int Nrows)
{
  __shared__ float fw[16 * 4 * 32 * 2];            // [kk][tile][lane][2]
  const int tid = threadIdx.x;
  for (int fi = tid; fi < 4096; fi += 256) {
    int j  = fi & 1, l = (fi >> 1) & 31;
    int t  = (fi >> 6) & 3, kk = fi >> 8;
    int K  = 4 * kk + 2 * (l >> 4) + j;            // B elem j: K = 2*half + j
    int col = 16 * t + (l & 15);
    fw[fi] = W[K * HDIM + col];
  }
  __syncthreads();

  const int lane = tid & 31, wave = tid >> 5;
  const int half = lane >> 4, ncol = lane & 15;
  const int rowbase = blockIdx.x * 128 + wave * 16;
  const int row = (rowbase + ncol < Nrows) ? (rowbase + ncol) : (Nrows - 1);
  const float* xr = Xin + (size_t)row * HDIM;
  const v2f* fwv = (const v2f*)fw;

  v8f acc0 = {}, acc1 = {}, acc2 = {}, acc3 = {};
  #pragma unroll 4
  for (int kk = 0; kk < 16; ++kk) {                // K = 64 in chunks of 4
    int k = kk * 4 + half * 2;                     // A: lane row = ncol, K = 2*half+j
    v2f a;  a[0] = xr[k];  a[1] = xr[k + 1];
    v2f b0 = fwv[(kk * 4 + 0) * 32 + lane];
    v2f b1 = fwv[(kk * 4 + 1) * 32 + lane];
    v2f b2 = fwv[(kk * 4 + 2) * 32 + lane];
    v2f b3 = fwv[(kk * 4 + 3) * 32 + lane];
    acc0 = wmma_f32x4(a, b0, acc0);
    acc1 = wmma_f32x4(a, b1, acc1);
    acc2 = wmma_f32x4(a, b2, acc2);
    acc3 = wmma_f32x4(a, b3, acc3);
  }

  #pragma unroll
  for (int e = 0; e < 8; ++e) {
    int r = rowbase + e + 8 * half;
    if (r < Nrows) {
      size_t o = (size_t)r * HDIM + ncol;
      Out[o]      = acc0[e];
      Out[o + 16] = acc1[e];
      Out[o + 32] = acc2[e];
      Out[o + 48] = acc3[e];
    }
  }
}

// ---------------------------------------------------------------------------
// Graph / BN helper kernels
// ---------------------------------------------------------------------------
__global__ void init_kernel(float* __restrict__ deg, float* __restrict__ bn, int Nn) {
  int i = blockIdx.x * 256 + threadIdx.x;
  if (i < Nn) deg[i] = 1.f;         // self-loop weight
  if (i < 128) bn[i] = 0.f;         // BN sum / sumsq
}

__global__ void edge_prep_aa(const int* __restrict__ idx, const float* __restrict__ len,
                             const float* __restrict__ we, const float* __restrict__ be,
                             float* __restrict__ wout, float* __restrict__ deg, int E) {
  int e = blockIdx.x * 256 + threadIdx.x;
  if (e >= E) return;
  float w = sigf(len[e] * we[0] + be[0]);
  wout[e] = w;
  unsafeAtomicAdd(&deg[idx[E + e]], w);          // dst = row 1
}

__global__ void edge_prep_al(const int* __restrict__ idx, const float* __restrict__ len,
                             const float* __restrict__ we, const float* __restrict__ be,
                             float* __restrict__ wout, float* __restrict__ deg, int E, int Aoff) {
  int e = blockIdx.x * 256 + threadIdx.x;
  if (e >= E) return;
  float w = sigf(len[e] * we[0] + be[0]);
  wout[e] = w;
  unsafeAtomicAdd(&deg[idx[E + e] + Aoff], w);   // fwd: dst = lane + A
  unsafeAtomicAdd(&deg[idx[e]], w);              // rev: dst = agent
}

__global__ void dinv_kernel(float* __restrict__ deg, int Nn) {
  int i = blockIdx.x * 256 + threadIdx.x;
  if (i < Nn) { float d = deg[i]; deg[i] = d > 0.f ? rsqrtf(d) : 0.f; }
}

// out[n] = b + hlin[n] * dinv[n]^2   (bias + self-loop term)
__global__ void agg_init_kernel(const float* __restrict__ hlin, const float* __restrict__ dinv,
                                const float* __restrict__ bias, float* __restrict__ outA,
                                float* __restrict__ outB, int Nn, int Asz) {
  long id = (long)blockIdx.x * 256 + threadIdx.x;
  if (id >= (long)Nn * HDIM) return;
  int n = (int)(id >> 6), ch = (int)(id & 63);
  float di = dinv[n];
  float v = bias[ch] + hlin[id] * di * di;
  float* dst = (n < Asz) ? (outA + (size_t)n * HDIM) : (outB + (size_t)(n - Asz) * HDIM);
  dst[ch] = v;
}

// 4 threads per edge, 16 channels each: out[dst] += h[src] * dinv[s]*w*dinv[d]
__global__ void scatter_kernel(const int* __restrict__ srcI, const int* __restrict__ dstI,
                               const float* __restrict__ w, const float* __restrict__ dinv,
                               const float* __restrict__ hsrc, float* __restrict__ outA,
                               float* __restrict__ outB, int E, int srcOff, int dstOff, int Asz) {
  long t = (long)blockIdx.x * 256 + threadIdx.x;
  int e = (int)(t >> 2), part = (int)(t & 3);
  if (e >= E) return;
  int s = srcI[e] + srcOff, d = dstI[e] + dstOff;
  const float* sp = hsrc + (size_t)s * HDIM + part * 16;
  float* dp = ((d < Asz) ? (outA + (size_t)d * HDIM) : (outB + (size_t)(d - Asz) * HDIM)) + part * 16;
  __builtin_prefetch(dp, 1, 3);                  // global_prefetch_b8: warm dst line
  float nw = dinv[s] * w[e] * dinv[d];
  #pragma unroll
  for (int i = 0; i < 16; ++i) unsafeAtomicAdd(dp + i, sp[i] * nw);
}

__global__ void bn_reduce_kernel(const float* __restrict__ h, float* __restrict__ bn, int Nn) {
  int ch = threadIdx.x & 63;
  int r0 = blockIdx.x * 4 + (threadIdx.x >> 6);
  float s = 0.f, sq = 0.f;
  for (int r = r0; r < Nn; r += gridDim.x * 4) {
    float v = h[(size_t)r * HDIM + ch];
    s += v; sq += v * v;
  }
  unsafeAtomicAdd(&bn[ch], s);
  unsafeAtomicAdd(&bn[64 + ch], sq);
}

__global__ void bn_final_kernel(float* __restrict__ bn, const float* __restrict__ gamma,
                                const float* __restrict__ beta, int Nn) {
  int ch = threadIdx.x;
  if (ch < 64) {
    float inv = 1.f / (float)Nn;
    float mu  = bn[ch] * inv;
    float var = bn[64 + ch] * inv - mu * mu;
    float sc  = gamma[ch] * rsqrtf(var + BN_EPS);
    bn[128 + ch] = sc;
    bn[192 + ch] = beta[ch] - mu * sc;
  }
}

__global__ void bn_apply_kernel(const float* __restrict__ h, const float* __restrict__ bn,
                                float* __restrict__ out, int Nn) {
  long id = (long)blockIdx.x * 256 + threadIdx.x;
  if (id >= (long)Nn * HDIM) return;
  int ch = (int)(id & 63);
  float v = fmaf(h[id], bn[128 + ch], bn[192 + ch]);
  out[id] = v > 0.f ? v : 0.f;
}

// ---------------------------------------------------------------------------
extern "C" void kernel_launch(void* const* d_in, const int* in_sizes, int n_in,
                              void* d_out, int out_size, void* d_ws, size_t ws_size,
                              hipStream_t stream)
{
  (void)in_sizes; (void)n_in; (void)out_size; (void)ws_size;

  const float* agent_hist = (const float*)d_in[0];
  const float* lane_nodes = (const float*)d_in[1];
  const int*   idx_aa     = (const int*)  d_in[2];
  const float* len_aa     = (const float*)d_in[3];
  const int*   idx_al     = (const int*)  d_in[4];
  const float* len_al     = (const float*)d_in[5];
  const float* Wih_a = (const float*)d_in[6];
  const float* Whh_a = (const float*)d_in[7];
  const float* b_a   = (const float*)d_in[8];
  const float* Wih_l = (const float*)d_in[9];
  const float* Whh_l = (const float*)d_in[10];
  const float* b_l   = (const float*)d_in[11];
  const float* w_e   = (const float*)d_in[12];
  const float* b_e   = (const float*)d_in[13];
  const float* W1    = (const float*)d_in[14];
  const float* b1    = (const float*)d_in[15];
  const float* gam1  = (const float*)d_in[16];
  const float* bet1  = (const float*)d_in[17];
  const float* W2    = (const float*)d_in[18];
  const float* b2    = (const float*)d_in[19];

  float* ws = (float*)d_ws;
  size_t o = 0;
  float* xfeat = ws + o;  o += (size_t)N_TOT * HDIM;   // node features / BN output
  float* hlin  = ws + o;  o += (size_t)N_TOT * HDIM;   // linear output (both convs)
  float* h1    = ws + o;  o += (size_t)N_TOT * HDIM;   // conv1 aggregation
  float* deg   = ws + o;  o += N_TOT;                  // degree -> dinv (in place)
  float* w_aa  = ws + o;  o += E_AA_N;
  float* w_al  = ws + o;  o += E_AL_N;
  float* bn    = ws + o;  o += 256;

  float* outp      = (float*)d_out;
  float* agent_emb = outp;
  float* xoutA     = outp + (size_t)A_TOT * HDIM;
  float* lane_emb  = outp + (size_t)2 * A_TOT * HDIM;
  float* xoutL     = outp + (size_t)2 * A_TOT * HDIM + (size_t)L_TOT * HDIM;

  const int LSTM_LDS = 49152 + 8 * 16 * 72 * 2;        // 67584 B (< 320KB/WGP)

  init_kernel<<<(N_TOT + 255) / 256, 256, 0, stream>>>(deg, bn, N_TOT);

  lstm_kernel<20, 5, false><<<(A_TOT + 127) / 128, 256, LSTM_LDS, stream>>>(
      agent_hist, Wih_a, Whh_a, b_a, agent_emb, xfeat, A_TOT);
  lstm_kernel<10, 2, true ><<<(L_TOT + 127) / 128, 256, LSTM_LDS, stream>>>(
      lane_nodes, Wih_l, Whh_l, b_l, lane_emb, xfeat + (size_t)A_TOT * HDIM, L_TOT);

  edge_prep_aa<<<(E_AA_N + 255) / 256, 256, 0, stream>>>(idx_aa, len_aa, w_e, b_e, w_aa, deg, E_AA_N);
  edge_prep_al<<<(E_AL_N + 255) / 256, 256, 0, stream>>>(idx_al, len_al, w_e, b_e, w_al, deg, E_AL_N, A_TOT);
  dinv_kernel<<<(N_TOT + 255) / 256, 256, 0, stream>>>(deg, N_TOT);

  const int gridLin = (N_TOT + 127) / 128;
  const int gridNH  = (int)(((size_t)N_TOT * HDIM + 255) / 256);

  // ---- conv1 ----
  linear_kernel<<<gridLin, 256, 0, stream>>>(xfeat, W1, hlin, N_TOT);
  agg_init_kernel<<<gridNH, 256, 0, stream>>>(hlin, deg, b1, h1, h1 + (size_t)A_TOT * HDIM, N_TOT, A_TOT);
  scatter_kernel<<<(E_AA_N * 4 + 255) / 256, 256, 0, stream>>>(
      idx_aa, idx_aa + E_AA_N, w_aa, deg, hlin, h1, h1 + (size_t)A_TOT * HDIM, E_AA_N, 0, 0, A_TOT);
  scatter_kernel<<<(E_AL_N * 4 + 255) / 256, 256, 0, stream>>>(
      idx_al, idx_al + E_AL_N, w_al, deg, hlin, h1, h1 + (size_t)A_TOT * HDIM, E_AL_N, 0, A_TOT, A_TOT);
  scatter_kernel<<<(E_AL_N * 4 + 255) / 256, 256, 0, stream>>>(
      idx_al + E_AL_N, idx_al, w_al, deg, hlin, h1, h1 + (size_t)A_TOT * HDIM, E_AL_N, A_TOT, 0, A_TOT);

  // ---- BN + ReLU ----
  bn_reduce_kernel<<<128, 256, 0, stream>>>(h1, bn, N_TOT);
  bn_final_kernel<<<1, 64, 0, stream>>>(bn, gam1, bet1, N_TOT);
  bn_apply_kernel<<<gridNH, 256, 0, stream>>>(h1, bn, xfeat, N_TOT);

  // ---- conv2 (writes straight into d_out regions) ----
  linear_kernel<<<gridLin, 256, 0, stream>>>(xfeat, W2, hlin, N_TOT);
  agg_init_kernel<<<gridNH, 256, 0, stream>>>(hlin, deg, b2, xoutA, xoutL, N_TOT, A_TOT);
  scatter_kernel<<<(E_AA_N * 4 + 255) / 256, 256, 0, stream>>>(
      idx_aa, idx_aa + E_AA_N, w_aa, deg, hlin, xoutA, xoutL, E_AA_N, 0, 0, A_TOT);
  scatter_kernel<<<(E_AL_N * 4 + 255) / 256, 256, 0, stream>>>(
      idx_al, idx_al + E_AL_N, w_al, deg, hlin, xoutA, xoutL, E_AL_N, 0, A_TOT, A_TOT);
  scatter_kernel<<<(E_AL_N * 4 + 255) / 256, 256, 0, stream>>>(
      idx_al + E_AL_N, idx_al, w_al, deg, hlin, xoutA, xoutL, E_AL_N, A_TOT, 0, A_TOT);
}
